// LoTDFMM_13537736917463
// MI455X (gfx1250) — matrix operand order repaired and checked by reading, not compiled, not run on hardware
//
#include <hip/hip_runtime.h>

typedef __attribute__((ext_vector_type(16))) _Float16 v16h;
typedef __attribute__((ext_vector_type(8)))  _Float16 v8h;
typedef __attribute__((ext_vector_type(8)))  float    v8f;

#define BATCH 8
#define ZDIM  128
#define WID   128
#define NF    4
#define NPTS0 32768            // 32^3
#define NPTS1 110592           // 48^3
#define TILE  128
#define TILES0 (NPTS0 / TILE)  // 256
#define TILES1 (NPTS1 / TILE)  // 864
#define LEVEL_STRIDE ((NPTS0 + NPTS1) * NF)  // 573440 floats per batch row
#define HPAD 136               // padded K stride in halfs (272B -> 4-bank shift/row)
#define W2ROWS 16              // W2 padded to 16 output channels for WMMA

// Truncating a flat (generic) shared-memory address to 32 bits yields the LDS
// byte offset (aperture bits live in addr[63:32] on gfx1250).
__device__ __forceinline__ unsigned lds_off(const void* p) {
    return (unsigned)(unsigned long long)p;
}

// ---------------------------------------------------------------------------
// Prep: FiLM modulation vectors (z @ M^T + mb) and f16 weight conversion.
// ---------------------------------------------------------------------------
__global__ __launch_bounds__(256) void lotd_prep_kernel(
    const float* __restrict__ z,
    const float* __restrict__ M0, const float* __restrict__ mb0,
    const float* __restrict__ M1, const float* __restrict__ mb1,
    const float* __restrict__ M2, const float* __restrict__ mb2,
    const float* __restrict__ W1, const float* __restrict__ W2,
    float* __restrict__ mod0, float* __restrict__ mod1, float* __restrict__ mod2,
    _Float16* __restrict__ W1h, _Float16* __restrict__ W2h)
{
    const int tid = threadIdx.x;

    // mod0 / mod1 : [8][256]
    for (int idx = tid; idx < BATCH * 2 * WID; idx += 256) {
        const int b = idx / (2 * WID), o = idx % (2 * WID);
        const float* zb = z + b * ZDIM;
        float a0 = mb0[o], a1 = mb1[o];
        for (int k = 0; k < ZDIM; ++k) {
            const float zv = zb[k];
            a0 += zv * M0[o * ZDIM + k];
            a1 += zv * M1[o * ZDIM + k];
        }
        mod0[idx] = a0;
        mod1[idx] = a1;
    }
    // mod2 : [8][8]
    for (int idx = tid; idx < BATCH * 2 * NF; idx += 256) {
        const int b = idx / (2 * NF), o = idx % (2 * NF);
        const float* zb = z + b * ZDIM;
        float a = mb2[o];
        for (int k = 0; k < ZDIM; ++k) a += zb[k] * M2[o * ZDIM + k];
        mod2[idx] = a;
    }
    // W1 -> f16, padded [128][HPAD]
    for (int idx = tid; idx < WID * WID; idx += 256) {
        const int c = idx / WID, k = idx % WID;
        W1h[c * HPAD + k] = (_Float16)W1[idx];
    }
    // W2 -> f16, padded [16][HPAD] (rows 4..15 and pad cols zeroed)
    for (int idx = tid; idx < W2ROWS * HPAD; idx += 256) {
        const int c = idx / HPAD, k = idx % HPAD;
        W2h[idx] = (c < NF && k < WID) ? (_Float16)W2[c * WID + k] : (_Float16)0.0f;
    }
}

// ---------------------------------------------------------------------------
// Fused FiLM-MLP: 1 workgroup = 1 batch element x 128 grid points, 8 waves.
// ---------------------------------------------------------------------------
__global__ __launch_bounds__(256) void lotd_fmm_kernel(
    const float* __restrict__ x0, const float* __restrict__ x1,
    const float* __restrict__ W0, const float* __restrict__ b0,
    const float* __restrict__ b1, const float* __restrict__ b2,
    const float* __restrict__ mod0, const float* __restrict__ mod1,
    const float* __restrict__ mod2,
    const _Float16* __restrict__ W1h, const _Float16* __restrict__ W2h,
    float* __restrict__ out)
{
    extern __shared__ char smem[];
    _Float16* Hs  = (_Float16*)smem;          // [128][HPAD] activations (f16)
    _Float16* W1s = Hs + TILE * HPAD;         // [128][HPAD] layer-1 weights
    _Float16* W2s = W1s + WID * HPAD;         // [16][HPAD]  layer-2 weights

    const int tid = threadIdx.x;
    const int bb  = blockIdx.y;
    const int tile = blockIdx.x;

    const float* xptr;
    int pbase;
    long outbase;
    if (tile < TILES0) {
        xptr = x0; pbase = tile * TILE;
        outbase = (long)bb * LEVEL_STRIDE + (long)pbase * NF;
    } else {
        xptr = x1; pbase = (tile - TILES0) * TILE;
        outbase = (long)bb * LEVEL_STRIDE + (long)NPTS0 * NF + (long)pbase * NF;
    }

    // ---- Async global->LDS staging of weights (overlaps with Stage A) -------
    {
        const unsigned w1l = lds_off(W1s);
        const unsigned long long w1g = (unsigned long long)W1h;
        for (int i = tid; i < (WID * HPAD * 2) / 16; i += 256) {
            const unsigned loff = w1l + i * 16;
            const unsigned goff = i * 16;
            asm volatile("global_load_async_to_lds_b128 %0, %1, %2"
                         :: "v"(loff), "v"(goff), "s"(w1g) : "memory");
        }
        const unsigned w2l = lds_off(W2s);
        const unsigned long long w2g = (unsigned long long)W2h;
        for (int i = tid; i < (W2ROWS * HPAD * 2) / 16; i += 256) {
            const unsigned loff = w2l + i * 16;
            const unsigned goff = i * 16;
            asm volatile("global_load_async_to_lds_b128 %0, %1, %2"
                         :: "v"(loff), "v"(goff), "s"(w2g) : "memory");
        }
    }

    // ---- Stage A: layer 0 (3 -> 128) + FiLM + ReLU, results to LDS as f16 ----
    {
        const int p = tid >> 1;                 // 2 threads per point
        const int cbase = (tid & 1) * 64;       // each covers 64 channels
        const int pg = pbase + p;
        const float xc = xptr[pg * 3 + 0];
        const float yc = xptr[pg * 3 + 1];
        const float zc = xptr[pg * 3 + 2];
        const float* sc = mod0 + bb * 2 * WID;  // scale
        const float* sh = sc + WID;             // shift
        for (int j0 = 0; j0 < 64; j0 += 8) {
            v8h tmp;
            #pragma unroll
            for (int j = 0; j < 8; ++j) {
                const int c = cbase + j0 + j;
                float v = xc * W0[c * 3 + 0] + yc * W0[c * 3 + 1] +
                          zc * W0[c * 3 + 2] + b0[c];
                v = v * (1.0f + sc[c]) + sh[c];
                v = v > 0.0f ? v : 0.0f;
                tmp[j] = (_Float16)v;
            }
            *(v8h*)(Hs + p * HPAD + cbase + j0) = tmp;
        }
    }
    asm volatile("s_wait_asynccnt 0x0" ::: "memory");
    __syncthreads();

    const int wave = tid >> 5;
    const int lane = tid & 31;
    const int lrow = lane & 15;
    const int lhi  = lane >> 4;

    // ---- Stage B: layer 1 (128 -> 128) via WMMA f32<-f16, FiLM + ReLU ------
    {
        // A fragments: rows wave*16+lrow, full K=128 (4 k-steps of 32).
        // 16-bit A layout: halfs 0..7 -> K = kb + lhi*8 .. +7,
        //                  halfs 8..15 -> K = kb + 16 + lhi*8 .. +7.
        v16h afrag[4];
        const _Float16* rp = Hs + (wave * 16 + lrow) * HPAD;
        #pragma unroll
        for (int ks = 0; ks < 4; ++ks) {
            const int kb = ks * 32 + lhi * 8;
            union { v16h v; v8h h[2]; } u;
            u.h[0] = *(const v8h*)(rp + kb);
            u.h[1] = *(const v8h*)(rp + kb + 16);
            afrag[ks] = u.v;
        }

        const float* sc = mod1 + bb * 2 * WID;
        const float* sh = sc + WID;

        #pragma unroll
        for (int ct = 0; ct < 8; ++ct) {
            const int c = ct * 16 + lrow;       // output channel for this lane
            v8f acc = {};
            #pragma unroll
            for (int ks = 0; ks < 4; ++ks) {
                // B layout: lane holds column n = lane%16,
                // halfs 0..7 -> K = ks*32 + lhi*8 .. +7, halfs 8..15 -> +16 (B[k][n] = W1[n][k]).
                const int kb = ks * 32 + lhi * 8;
                const _Float16* wp = W1s + c * HPAD + kb;
                union { v16h v; v8h h[2]; } ub;
                ub.h[0] = *(const v8h*)wp;
                ub.h[1] = *(const v8h*)(wp + 16);
                acc = __builtin_amdgcn_wmma_f32_16x16x32_f16(
                        false, afrag[ks], false, ub.v,
                        (short)0, acc, false, false);
            }
            // Epilogue: bias + FiLM + ReLU; lane's column (channel) is uniform
            // across the 8 accumulator rows -> one scale/shift/bias per lane.
            const float s = 1.0f + sc[c];
            const float t = sh[c];
            const float bias = b1[c];
            #pragma unroll
            for (int i = 0; i < 8; ++i) {
                float v = (acc[i] + bias) * s + t;
                v = v > 0.0f ? v : 0.0f;
                const int row = wave * 16 + i + lhi * 8;   // C/D row mapping
                Hs[row * HPAD + c] = (_Float16)v;          // in-place: own rows
            }
        }
    }
    __syncthreads();

    // ---- Stage C: layer 2 (128 -> 4, N padded to 16) via WMMA + FiLM -------
    {
        v16h afrag[4];
        const _Float16* rp = Hs + (wave * 16 + lrow) * HPAD;
        #pragma unroll
        for (int ks = 0; ks < 4; ++ks) {
            const int kb = ks * 32 + lhi * 8;
            union { v16h v; v8h h[2]; } u;
            u.h[0] = *(const v8h*)(rp + kb);
            u.h[1] = *(const v8h*)(rp + kb + 16);
            afrag[ks] = u.v;
        }
        v8f acc = {};
        #pragma unroll
        for (int ks = 0; ks < 4; ++ks) {
            const int kb = ks * 32 + lhi * 8;
            const _Float16* wp = W2s + lrow * HPAD + kb;   // B[k][n] = W2[n][k]
            union { v16h v; v8h h[2]; } ub;
            ub.h[0] = *(const v8h*)wp;
            ub.h[1] = *(const v8h*)(wp + 16);
            acc = __builtin_amdgcn_wmma_f32_16x16x32_f16(
                    false, afrag[ks], false, ub.v,
                    (short)0, acc, false, false);
        }
        // Only D columns 0..3 are real output features; stage them to LDS
        // (reusing the W1s region) for a coalesced float4 global store.
        float* Fout = (float*)W1s;                         // [128][4] floats
        const float* m2 = mod2 + bb * 2 * NF;
        const int f = (lrow < NF) ? lrow : 0;
        const float s2 = 1.0f + m2[f];
        const float t2 = m2[NF + f];
        const float bias2 = b2[f];
        if (lrow < NF) {
            #pragma unroll
            for (int i = 0; i < 8; ++i) {
                const int row = wave * 16 + i + lhi * 8;
                Fout[row * NF + lrow] = (acc[i] + bias2) * s2 + t2;
            }
        }
    }
    __syncthreads();

    // ---- Coalesced output: 512 floats = 128 float4 stores ------------------
    if (tid < TILE * NF / 4) {
        ((float4*)(out + outbase))[tid] = ((const float4*)(const void*)W1s)[tid];
    }
}

// ---------------------------------------------------------------------------
extern "C" void kernel_launch(void* const* d_in, const int* in_sizes, int n_in,
                              void* d_out, int out_size, void* d_ws, size_t ws_size,
                              hipStream_t stream)
{
    const float* z   = (const float*)d_in[0];
    const float* x0  = (const float*)d_in[1];
    const float* x1  = (const float*)d_in[2];
    const float* W0  = (const float*)d_in[3];
    const float* b0  = (const float*)d_in[4];
    const float* M0  = (const float*)d_in[5];
    const float* mb0 = (const float*)d_in[6];
    const float* W1  = (const float*)d_in[7];
    const float* b1  = (const float*)d_in[8];
    const float* M1  = (const float*)d_in[9];
    const float* mb1 = (const float*)d_in[10];
    const float* W2  = (const float*)d_in[11];
    const float* b2  = (const float*)d_in[12];
    const float* M2  = (const float*)d_in[13];
    const float* mb2 = (const float*)d_in[14];
    float* out = (float*)d_out;

    // Workspace layout
    float* mod0 = (float*)d_ws;                    // [8][256]
    float* mod1 = mod0 + BATCH * 2 * WID;          // [8][256]
    float* mod2 = mod1 + BATCH * 2 * WID;          // [8][8]
    _Float16* W1h = (_Float16*)(mod2 + BATCH * 2 * NF);  // [128][HPAD]
    _Float16* W2h = W1h + WID * HPAD;                    // [16][HPAD]

    lotd_prep_kernel<<<1, 256, 0, stream>>>(z, M0, mb0, M1, mb1, M2, mb2,
                                            W1, W2, mod0, mod1, mod2, W1h, W2h);

    const size_t smem = (size_t)(TILE * HPAD + WID * HPAD + W2ROWS * HPAD) * sizeof(_Float16);
    dim3 grid(TILES0 + TILES1, BATCH);
    lotd_fmm_kernel<<<grid, 256, smem, stream>>>(x0, x1, W0, b0, b1, b2,
                                                 mod0, mod1, mod2, W1h, W2h, out);
}
